// BlockDiagMNISTConvertible_45552423141699
// MI455X (gfx1250) — matrix operand order repaired and checked by reading, not compiled, not run on hardware
//
#include <hip/hip_runtime.h>
#include <hip/hip_bf16.h>

typedef __attribute__((ext_vector_type(16))) _Float16 v16h;
typedef __attribute__((ext_vector_type(8)))  _Float16 v8h;
typedef __attribute__((ext_vector_type(4)))  _Float16 v4h;
typedef __attribute__((ext_vector_type(8)))  float    v8f;

#define HIDDEN     4096
#define NBLK       128
#define BLK        32
#define IN_DIM     784
#define INP        800          // IN_DIM padded to multiple of 32 (zeros)
#define XS_STRIDE  816          // LDS x-tile row stride in halfs (16B aligned, conflict-spread)
#define OUT_DIM    10
#define BATCH      32768
#define ROWS_PER_WG 64
#define WAVES      4
#define SCR_STRIDE 40           // per-wave scratch row stride in halfs (80B: 16-lane conflict-free)

// ---------------------------------------------------------------------------
// A-matrix fragment (16x32 f16, M x K) from LDS row-major tile.
// Layout: lane L<16 holds row M=L, K=0..15;
//         lane L+16 holds row M=L, K=16..31 (matches B fragments).
// => two ds_load_b128 at byte offsets 32*half and 32*half+16 within the row.
__device__ __forceinline__ v16h lds_loadA(const _Float16* rowbase, int stride,
                                          int k0, int lane) {
    const int m    = lane & 15;
    const int half = lane >> 4;
    const _Float16* p = rowbase + m * stride + k0 + half * 16;  // +32B per half
    v8h lo = *(const v8h*)p;                                    // K k0+.. 8 halfs
    v8h hi = *(const v8h*)(p + 8);                              // +16B
    return __builtin_shufflevector(lo, hi, 0,1,2,3,4,5,6,7,8,9,10,11,12,13,14,15);
}

// B-matrix fragment (32x16 f16, K x N) from global, row-major [N][K] weights.
// Layout used here (contiguous k, same as A): lane L<16 holds col N=L, K=0..15; lane L+16 holds col N=L-16, K=16..31.
// => 32 consecutive bytes per lane at (rowN)*stride + k0 + 16*half halfs.
__device__ __forceinline__ v16h glb_loadB(const _Float16* colbase /* + N*stride */,
                                          int k0, int half) {
    return *(const v16h*)(colbase + k0 + half * 16);
}

__device__ __forceinline__ v8f wmma_f16(v16h a, v16h b, v8f c) {
    return __builtin_amdgcn_wmma_f32_16x16x32_f16(false, a, false, b,
                                                  (short)0, c, false, false);
}

// ---------------------------------------------------------------------------
// Weight prep: f32 -> f16 (+ zero padding). One flat elementwise kernel.
__global__ void prep_weights(const float* __restrict__ W1,
                             const float* __restrict__ Wb,
                             const float* __restrict__ W3,
                             _Float16* __restrict__ w1f,   // [4096][800]
                             _Float16* __restrict__ wbf,   // [128][32][32] ([n][o][k])
                             _Float16* __restrict__ w3f)   // [16][4096], rows>=10 zero
{
    const int N1 = HIDDEN * INP;
    const int N2 = NBLK * BLK * BLK;
    const int N3 = 16 * HIDDEN;
    int idx = blockIdx.x * blockDim.x + threadIdx.x;
    if (idx < N1) {
        int h = idx / INP, k = idx - h * INP;
        w1f[idx] = (k < IN_DIM) ? (_Float16)W1[(size_t)h * IN_DIM + k] : (_Float16)0.f;
    } else if (idx < N1 + N2) {
        int j = idx - N1;
        wbf[j] = (_Float16)Wb[j];
    } else if (idx < N1 + N2 + N3) {
        int j = idx - N1 - N2;
        int o = j / HIDDEN, k = j - o * HIDDEN;
        w3f[j] = (o < OUT_DIM) ? (_Float16)W3[(size_t)o * HIDDEN + k] : (_Float16)0.f;
    }
}

// ---------------------------------------------------------------------------
// Fused: relu(x@W1^T+b1) -> per-block 32x32 -> relu(+bb) -> @W3^T + b3.
// 4 waves * 16 rows = 64 batch rows per workgroup; x tile staged once in LDS.
__global__ __launch_bounds__(WAVES * 32)
void fused_mlp(const float* __restrict__ x,
               const float* __restrict__ pb1,
               const float* __restrict__ pbb,
               const float* __restrict__ pb3,
               const _Float16* __restrict__ w1,   // [4096][800] f16
               const _Float16* __restrict__ wb,   // [128][32][32] f16
               const _Float16* __restrict__ w3,   // [16][4096] f16
               float* __restrict__ out)           // [32768][10] f32
{
    extern __shared__ _Float16 smem[];
    _Float16* s_x   = smem;                                   // 64*816 halfs
    _Float16* s_scr = smem + ROWS_PER_WG * XS_STRIDE;         // 4*16*40 halfs

    const int tid    = threadIdx.x;
    const int lane   = tid & 31;
    const int wave   = tid >> 5;
    const int col    = lane & 15;
    const int half   = lane >> 4;
    const int wgRow0 = blockIdx.x * ROWS_PER_WG;

    // ---- stage x tile into LDS as f16 (cols >= 784 zero-padded) ----
    for (int i = tid; i < ROWS_PER_WG * (XS_STRIDE / 4); i += WAVES * 32) {
        int r  = i / (XS_STRIDE / 4);
        int c4 = (i - r * (XS_STRIDE / 4)) * 4;
        v4h hv;
        if (c4 + 3 < IN_DIM) {
            const float4 f = *(const float4*)(x + (size_t)(wgRow0 + r) * IN_DIM + c4);
            hv[0] = (_Float16)f.x; hv[1] = (_Float16)f.y;
            hv[2] = (_Float16)f.z; hv[3] = (_Float16)f.w;
        } else {
            hv[0] = (_Float16)0.f; hv[1] = (_Float16)0.f;
            hv[2] = (_Float16)0.f; hv[3] = (_Float16)0.f;
        }
        *(v4h*)&s_x[r * XS_STRIDE + c4] = hv;
    }
    __syncthreads();

    const _Float16* s_rows = s_x + wave * 16 * XS_STRIDE;      // this wave's 16 rows
    _Float16*       scr    = s_scr + wave * 16 * SCR_STRIDE;   // private scratch

    v8f accOut = {};  // persistent layer-3 accumulator (cols 0..15, rows of this tile)

    for (int n = 0; n < NBLK; ++n) {
        // -------- layer 1: h16x32 = x16x800 @ W1_block^T, K tiled by 32 --------
        const _Float16* w1c0 = w1 + (size_t)(n * BLK + col) * INP;  // col N -> W1 row
        const _Float16* w1c1 = w1c0 + (size_t)16 * INP;
        if (n + 1 < NBLK) __builtin_prefetch(w1c0 + (size_t)BLK * INP, 0, 1);

        v8f acc0 = {}, acc1 = {};
        #pragma unroll 5
        for (int k0 = 0; k0 < INP; k0 += 32) {
            v16h aF = lds_loadA(s_rows, XS_STRIDE, k0, lane);
            v16h b0 = glb_loadB(w1c0, k0, half);
            v16h b1 = glb_loadB(w1c1, k0, half);
            acc0 = wmma_f16(aF, b0, acc0);
            acc1 = wmma_f16(aF, b1, acc1);
        }

        // bias + relu in C/D layout (lane col is constant), spill to scratch as f16
        {
            const float g0 = pb1[n * BLK + col];
            const float g1 = pb1[n * BLK + 16 + col];
            #pragma unroll
            for (int r = 0; r < 8; ++r) {
                int row = r + 8 * half;
                scr[row * SCR_STRIDE + col]      = (_Float16)fmaxf(acc0[r] + g0, 0.f);
                scr[row * SCR_STRIDE + col + 16] = (_Float16)fmaxf(acc1[r] + g1, 0.f);
            }
        }

        // -------- layer 2: block-diagonal 32x32 (single K=32 WMMA pair) --------
        {
            v16h aF  = lds_loadA(scr, SCR_STRIDE, 0, lane);
            const _Float16* wbn = wb + (size_t)n * (BLK * BLK);   // [o][k], k contiguous
            v16h bF0 = *(const v16h*)(wbn + (size_t)col * BLK + half * 16);
            v16h bF1 = *(const v16h*)(wbn + (size_t)(col + 16) * BLK + half * 16);
            v8f z = {};
            v8f y0 = wmma_f16(aF, bF0, z);
            v8f y1 = wmma_f16(aF, bF1, z);

            const float g0 = pbb[n * BLK + col];
            const float g1 = pbb[n * BLK + 16 + col];
            #pragma unroll
            for (int r = 0; r < 8; ++r) {
                int row = r + 8 * half;
                scr[row * SCR_STRIDE + col]      = (_Float16)fmaxf(y0[r] + g0, 0.f);
                scr[row * SCR_STRIDE + col + 16] = (_Float16)fmaxf(y1[r] + g1, 0.f);
            }
        }

        // -------- layer 3 partial: h2_block @ W3_block^T, accumulate --------
        {
            v16h aF = lds_loadA(scr, SCR_STRIDE, 0, lane);
            v16h bF = *(const v16h*)(w3 + (size_t)col * HIDDEN + n * BLK + half * 16);
            accOut = wmma_f16(aF, bF, accOut);
        }
    }

    // ---- epilogue: + b3, store [rows x 10] f32 ----
    if (col < OUT_DIM) {
        const float g = pb3[col];
        #pragma unroll
        for (int r = 0; r < 8; ++r) {
            int row = wgRow0 + wave * 16 + r + 8 * half;
            out[(size_t)row * OUT_DIM + col] = accOut[r] + g;
        }
    }
}

// ---------------------------------------------------------------------------
extern "C" void kernel_launch(void* const* d_in, const int* in_sizes, int n_in,
                              void* d_out, int out_size, void* d_ws, size_t ws_size,
                              hipStream_t stream) {
    const float* x  = (const float*)d_in[0];
    const float* W1 = (const float*)d_in[1];
    const float* b1 = (const float*)d_in[2];
    const float* Wb = (const float*)d_in[3];
    const float* bb = (const float*)d_in[4];
    const float* W3 = (const float*)d_in[5];
    const float* b3 = (const float*)d_in[6];
    float* out = (float*)d_out;

    _Float16* w1f = (_Float16*)d_ws;                 // 4096*800 halfs
    _Float16* wbf = w1f + (size_t)HIDDEN * INP;      // 128*32*32 halfs
    _Float16* w3f = wbf + (size_t)NBLK * BLK * BLK;  // 16*4096 halfs

    const int prepN = HIDDEN * INP + NBLK * BLK * BLK + 16 * HIDDEN;
    prep_weights<<<(prepN + 255) / 256, 256, 0, stream>>>(W1, Wb, W3, w1f, wbf, w3f);

    const size_t shmem =
        (size_t)(ROWS_PER_WG * XS_STRIDE + WAVES * 16 * SCR_STRIDE) * sizeof(_Float16);
    fused_mlp<<<BATCH / ROWS_PER_WG, WAVES * 32, shmem, stream>>>(
        x, b1, bb, b3, w1f, wbf, w3f, out);
}
